// TokenSelection_3152505995575
// MI455X (gfx1250) — compile-verified, matrix operand order checked by reading, not run on hardware
//
#include <hip/hip_runtime.h>
#include <hip/hip_bf16.h>
#include <math.h>

// ---------------------------------------------------------------------------
// TokenSelection, algebraically reduced:
//   * softmax dropped (monotonic under top_k)
//   * bk dropped (adds per-(b,q) constant across keys -> ranking invariant)
//   * keys@Wk.T (1.8e11 FLOPs) folded into QK = (Gstack@Wq.T + bq) @ Wk
//     (two 768x768x768 GEMMs, done with v_wmma_f32_16x16x32_bf16)
//   * Wk transposed during bf16 conversion so BOTH GEMMs use the contiguous
//     B[N][K] operand path (global_load_b128, no strided b16 gathers)
//   * scores = dot(patch, {own_global, QK[q1], QK[q2]})  -- memory bound
//   * top-k via LDS bitonic sort -> threshold -> mask
//   * apply: out = patch * union(mask)   (float4 streaming)
// ---------------------------------------------------------------------------

typedef __attribute__((ext_vector_type(16))) __bf16 v16bf;
typedef __attribute__((ext_vector_type(8)))  float  v8f;

#define DIMD 768
#define NB   256
#define NP   196
#define K1V  112
#define K2V  224

// ---- pack three [256,768] global vectors into one bf16 [768,768] matrix ----
__global__ void pack_globals_bf16(const float* __restrict__ rgb,
                                  const float* __restrict__ nir,
                                  const float* __restrict__ tir,
                                  __bf16* __restrict__ Gs) {
  int idx = blockIdx.x * blockDim.x + threadIdx.x;
  if (idx >= DIMD * DIMD) return;
  int r = idx / DIMD, e = idx % DIMD;
  int q = r >> 8, b = r & 255;
  const float* src = (q == 0) ? rgb : (q == 1) ? nir : tir;
  Gs[idx] = (__bf16)src[(size_t)b * DIMD + e];
}

__global__ void convert_f32_bf16(const float* __restrict__ in,
                                 __bf16* __restrict__ out, int n) {
  int idx = blockIdx.x * blockDim.x + threadIdx.x;
  if (idx < n) out[idx] = (__bf16)in[idx];
}

// ---- LDS-tiled transpose with f32->bf16 conversion: out[d][e] = in[e][d] --
__global__ void transpose_f32_bf16(const float* __restrict__ in,
                                   __bf16* __restrict__ out) {
  __shared__ float tile[32][33];                 // +1 pad: no bank conflicts
  const int T = DIMD / 32;                       // 24 tiles per dim
  int bx = blockIdx.x % T, by = blockIdx.x / T;
  int tx = threadIdx.x, ty = threadIdx.y;        // (32, 8)
  for (int i = ty; i < 32; i += 8)
    tile[i][tx] = in[(size_t)(by * 32 + i) * DIMD + bx * 32 + tx];
  __syncthreads();
  for (int i = ty; i < 32; i += 8)
    out[(size_t)(bx * 32 + i) * DIMD + by * 32 + tx] = (__bf16)tile[tx][i];
}

// ---- 768x768x768 bf16 WMMA GEMM, one 16x16 tile per wave32 ----------------
// B is stored [N][K] (row-major by N -> contiguous in K for every lane)
template <bool STORE_BF16, bool ADD_BIAS>
__global__ void wmma_gemm768(const __bf16* __restrict__ A,
                             const __bf16* __restrict__ B,
                             const float* __restrict__ bias,
                             float* __restrict__ Cf,
                             __bf16* __restrict__ Cb) {
  const int TPW = 48;  // 768/16 tiles per dimension
  int wave = (blockIdx.x * blockDim.x + threadIdx.x) >> 5;  // 0..2303 exactly
  int lane = threadIdx.x & 31;
  int tm = wave / TPW, tn = wave % TPW;

  int mrow  = tm * 16 + (lane & 15);      // A row owned by this lane
  int ncol  = tn * 16 + (lane & 15);      // B/C column owned by this lane
  int kbase = (lane >> 4) << 3;           // 0 for lanes 0-15, 8 for 16-31

  v8f acc = {0.f, 0.f, 0.f, 0.f, 0.f, 0.f, 0.f, 0.f};

  for (int k0 = 0; k0 < DIMD; k0 += 32) {
    v16bf av, bv;
    // 16-bit A 16x32 layout: lane holds K = kbase+{0..7} and kbase+16+{0..7}
    const __bf16* ap = A + (size_t)mrow * DIMD + k0 + kbase;
    const __bf16* bp = B + (size_t)ncol * DIMD + k0 + kbase;
#pragma unroll
    for (int i = 0; i < 8; ++i) {
      av[i] = ap[i]; av[i + 8] = ap[16 + i];
      bv[i] = bp[i]; bv[i + 8] = bp[16 + i];
    }
    acc = __builtin_amdgcn_wmma_f32_16x16x32_bf16(
        /*neg_a=*/false, av, /*neg_b=*/false, bv,
        /*c_mod=*/(short)0, acc, /*reuse_a=*/false, /*reuse_b=*/false);
  }

  // C/D layout: N = lane%16, M = vgpr_j + 8*(lane>=16)
  int rowbase = tm * 16 + ((lane >> 4) << 3);
  float bcol = ADD_BIAS ? bias[ncol] : 0.f;
#pragma unroll
  for (int j = 0; j < 8; ++j) {
    float v = acc[j] + bcol;
    size_t off = (size_t)(rowbase + j) * DIMD + ncol;
    if (STORE_BF16) Cb[off] = (__bf16)v;
    else            Cf[off] = v;
  }
}

__device__ __forceinline__ int cross_slot(int m, int q) {
  int other = 3 - q - m;                 // the third modality
  return (m < other) ? 0 : NP;           // concat order: ascending modality
}

// ---- per-(modality,batch): 3 dots per patch: own global + two QK rows -----
__global__ void scores_kernel(const float* __restrict__ rgb,
                              const float* __restrict__ nir,
                              const float* __restrict__ tir,
                              const float* __restrict__ rgbg,
                              const float* __restrict__ nirg,
                              const float* __restrict__ tirg,
                              const float* __restrict__ QK,
                              float* __restrict__ intra,
                              float* __restrict__ cross) {
  int bm = blockIdx.x;                   // 0..767
  int m = bm / NB, b = bm % NB;
  const float* patches = (m == 0) ? rgb : (m == 1) ? nir : tir;
  const float* g = ((m == 0) ? rgbg : (m == 1) ? nirg : tirg) + (size_t)b * DIMD;
  int q1 = (m == 0) ? 1 : 0;
  int q2 = (m == 2) ? 1 : 2;
  const float* qa = QK + (size_t)(q1 * NB + b) * DIMD;
  const float* qb = QK + (size_t)(q2 * NB + b) * DIMD;
  int o1 = cross_slot(m, q1), o2 = cross_slot(m, q2);

  __shared__ float4 sg[DIMD / 4], sa[DIMD / 4], sb[DIMD / 4];
  int t = threadIdx.x;
  if (t < DIMD / 4) {
    sg[t] = ((const float4*)g)[t];
    sa[t] = ((const float4*)qa)[t];
    sb[t] = ((const float4*)qb)[t];
  }
  __syncthreads();

  int wave = t >> 5, lane = t & 31;
  for (int n = wave; n < NP; n += 8) {
    const float4* row = (const float4*)(patches + ((size_t)b * NP + n) * DIMD);
    float ag = 0.f, aa = 0.f, ab = 0.f;
#pragma unroll
    for (int i = 0; i < 6; ++i) {
      int idx = i * 32 + lane;
      float4 x = row[idx];
      float4 vg = sg[idx], va = sa[idx], vb = sb[idx];
      ag += x.x * vg.x + x.y * vg.y + x.z * vg.z + x.w * vg.w;
      aa += x.x * va.x + x.y * va.y + x.z * va.z + x.w * va.w;
      ab += x.x * vb.x + x.y * vb.y + x.z * vb.z + x.w * vb.w;
    }
#pragma unroll
    for (int off = 16; off > 0; off >>= 1) {
      ag += __shfl_xor(ag, off, 32);
      aa += __shfl_xor(aa, off, 32);
      ab += __shfl_xor(ab, off, 32);
    }
    if (lane == 0) {
      intra[((size_t)m * NB + b) * NP + n] = ag;
      cross[((size_t)(q1 * NB + b)) * (2 * NP) + o1 + n] = aa;
      cross[((size_t)(q2 * NB + b)) * (2 * NP) + o2 + n] = ab;
    }
  }
}

// ---- intra top-K1 of 196 via 256-wide bitonic sort ------------------------
__global__ void topk_intra(const float* __restrict__ intra,
                           float* __restrict__ maskI) {
  int mb = blockIdx.x;                   // m*256+b
  __shared__ float s[256];
  int t = threadIdx.x;
  float v = (t < NP) ? intra[(size_t)mb * NP + t] : -INFINITY;
  s[t] = v;
  __syncthreads();
  for (int k = 2; k <= 256; k <<= 1)
    for (int j = k >> 1; j > 0; j >>= 1) {
      int ixj = t ^ j;
      if (ixj > t) {
        float a = s[t], c = s[ixj];
        bool dsc = ((t & k) == 0);       // descending overall
        if (dsc ? (a < c) : (a > c)) { s[t] = c; s[ixj] = a; }
      }
      __syncthreads();
    }
  float thr = s[K1V - 1];
  if (t < NP) maskI[(size_t)mb * NP + t] = (v >= thr) ? 1.f : 0.f;
}

// ---- cross top-K2 of 392 via 512-wide bitonic sort (2 elems/thread) -------
__global__ void topk_cross(const float* __restrict__ cross,
                           float* __restrict__ maskX) {
  int qb = blockIdx.x;                   // q*256+b
  __shared__ float s[512];
  int t = threadIdx.x;
  float v0 = (t < 2 * NP) ? cross[(size_t)qb * (2 * NP) + t] : -INFINITY;
  float v1 = (t + 256 < 2 * NP) ? cross[(size_t)qb * (2 * NP) + t + 256] : -INFINITY;
  s[t] = v0; s[t + 256] = v1;
  __syncthreads();
  for (int k = 2; k <= 512; k <<= 1)
    for (int j = k >> 1; j > 0; j >>= 1) {
      for (int base = 0; base < 512; base += 256) {
        int i = base + t;
        int ixj = i ^ j;
        if (ixj > i) {
          float a = s[i], c = s[ixj];
          bool dsc = ((i & k) == 0);
          if (dsc ? (a < c) : (a > c)) { s[i] = c; s[ixj] = a; }
        }
      }
      __syncthreads();
    }
  float thr = s[K2V - 1];
  if (t < 2 * NP)
    maskX[(size_t)qb * (2 * NP) + t] = (v0 >= thr) ? 1.f : 0.f;
  if (t + 256 < 2 * NP)
    maskX[(size_t)qb * (2 * NP) + t + 256] = (v1 >= thr) ? 1.f : 0.f;
}

// ---- union masks, write out = patch * mask (float4 streaming) -------------
__global__ void apply_kernel(const float* __restrict__ rgb,
                             const float* __restrict__ nir,
                             const float* __restrict__ tir,
                             const float* __restrict__ maskI,
                             const float* __restrict__ maskX,
                             float* __restrict__ out) {
  int bm = blockIdx.x;
  int m = bm / NB, b = bm % NB;
  const float* patches = (m == 0) ? rgb : (m == 1) ? nir : tir;
  int q1 = (m == 0) ? 1 : 0;
  int q2 = (m == 2) ? 1 : 2;
  int o1 = cross_slot(m, q1), o2 = cross_slot(m, q2);

  __shared__ float msk[NP];
  int t = threadIdx.x;
  if (t < NP) {
    float f = maskI[((size_t)m * NB + b) * NP + t] +
              maskX[((size_t)(q1 * NB + b)) * (2 * NP) + o1 + t] +
              maskX[((size_t)(q2 * NB + b)) * (2 * NP) + o2 + t];
    msk[t] = (f > 0.f) ? 1.f : 0.f;
  }
  __syncthreads();

  const float4* in4 = (const float4*)(patches + (size_t)b * NP * DIMD);
  float4* out4 = (float4*)(out + ((size_t)m * NB * NP + (size_t)b * NP) * DIMD);
  const int NV = NP * (DIMD / 4);        // 196*192 float4 per (m,b)
  for (int idx = t; idx < NV; idx += 256) {
    int n = idx / (DIMD / 4);
    float sc = msk[n];
    float4 x = in4[idx];
    x.x *= sc; x.y *= sc; x.z *= sc; x.w *= sc;
    out4[idx] = x;
  }
}

extern "C" void kernel_launch(void* const* d_in, const int* in_sizes, int n_in,
                              void* d_out, int out_size, void* d_ws, size_t ws_size,
                              hipStream_t stream) {
  const float* rgb  = (const float*)d_in[0];
  const float* nir  = (const float*)d_in[1];
  const float* tir  = (const float*)d_in[2];
  const float* rgbg = (const float*)d_in[3];
  const float* nirg = (const float*)d_in[4];
  const float* tirg = (const float*)d_in[5];
  const float* Wq   = (const float*)d_in[6];
  const float* bq   = (const float*)d_in[7];
  const float* Wk   = (const float*)d_in[8];
  // d_in[9] = bk: intentionally unused (adds a per-(b,q) constant across all
  // keys -> ranking under top_k is invariant).

  const size_t MAT = (size_t)DIMD * DIMD;           // 589824
  __bf16* Gs   = (__bf16*)d_ws;
  __bf16* Wqb  = Gs + MAT;
  __bf16* Wkbt = Wqb + MAT;                          // Wk^T in bf16
  __bf16* Qb   = Wkbt + MAT;
  float*  QK    = (float*)(Qb + MAT);
  float*  intra = QK + MAT;                          // 3*256*196
  float*  cross = intra + 3 * NB * NP;               // 3*256*392
  float*  maskI = cross + 3 * NB * 2 * NP;
  float*  maskX = maskI + 3 * NB * NP;
  // total ~10.2 MB of d_ws

  float* out = (float*)d_out;

  // 1. pack/convert to bf16 (Wk transposed so GEMM2's B operand is [N][K])
  pack_globals_bf16<<<(MAT + 255) / 256, 256, 0, stream>>>(rgbg, nirg, tirg, Gs);
  convert_f32_bf16<<<(MAT + 255) / 256, 256, 0, stream>>>(Wq, Wqb, (int)MAT);
  transpose_f32_bf16<<<(DIMD / 32) * (DIMD / 32), dim3(32, 8), 0, stream>>>(Wk, Wkbt);

  // 2. Q = Gs @ Wq^T + bq   (B[N=d][K=e] = Wq[d][e], already row-major)
  wmma_gemm768<true, true><<<288, 256, 0, stream>>>(Gs, Wqb, bq, nullptr, Qb);
  // 3. QK = Q @ Wk          (B[N=d][K=e] = Wk[e][d] = Wkbt[d][e])
  wmma_gemm768<false, false><<<288, 256, 0, stream>>>(Qb, Wkbt, nullptr, QK, nullptr);

  // 4. per-patch scores (intra + two cross queries)
  scores_kernel<<<3 * NB, 256, 0, stream>>>(rgb, nir, tir, rgbg, nirg, tirg,
                                            QK, intra, cross);
  // 5. top-k -> masks
  topk_intra<<<3 * NB, 256, 0, stream>>>(intra, maskI);
  topk_cross<<<3 * NB, 256, 0, stream>>>(cross, maskX);
  // 6. union + apply
  apply_kernel<<<3 * NB, 256, 0, stream>>>(rgb, nir, tir, maskI, maskX, out);
}
